// MPNN_49014166782074
// MI455X (gfx1250) — matrix operand order repaired and checked by reading, not compiled, not run on hardware
//
#include <hip/hip_runtime.h>
#include <math.h>

#define NN 20000
#define EE 100000
#define MM 1000
#define RR 250
#define DD 32
#define DHH 512

typedef __bf16 bf16;
typedef __attribute__((ext_vector_type(16))) __bf16 v16bf;
typedef __attribute__((ext_vector_type(8)))  float  v8f;

// Packed fragment-buffer element counts (bf16), in allocation order.
#define WPF_N   2048          // Wp:    2 K-chunks * 2 col-tiles
#define W2F_N   17408         // Wb(16 chunks) + bb(1 chunk), 2 col-tiles each
#define WIHF_N  3072          // W_ih^T: 1 chunk * 6 tiles
#define WHHF_N  3072
#define WLIHF_N 32768         // Wl_ih^T: 4 chunks * 16 tiles
#define WLHHF_N 16384         // Wl_hh^T: 2 chunks * 16 tiles
#define WSF_N   65536         // Ws: 4 chunks * 32 tiles

__device__ __forceinline__ v8f wmma_bf16(v16bf a, v16bf b, v8f c) {
  // D = A(16x32 bf16) * B(32x16 bf16) + C(16x16 f32)
  return __builtin_amdgcn_wmma_f32_16x16x32_bf16(false, a, false, b, (short)0, c,
                                                 false, false);
}
// A-fragment K index for element j (contiguous order k = 16*half + j, matching packed B)
__device__ __forceinline__ int kA(int j, int half) { return 16 * half + j; }
__device__ __forceinline__ float sigm(float x) { return 1.0f / (1.0f + __expf(-x)); }

// B fragment, pre-packed: base[((frag*32)+lane)*16 .. +15] -> one v16bf (2x b128)
__device__ __forceinline__ v16bf load_frag(const bf16* __restrict__ base, int frag, int lane) {
  return *(const v16bf*)(base + ((size_t)frag * 32 + lane) * 16);
}

// ---------------------------------------------------------------- weight prep
// Convert all GEMM B-operands to bf16 laid out per WMMA fragment, in this file's contiguous k order:
// element (frag, lane, j) holds B[k][col] with k = 16*(lane>>4)+j (+32*t),
// col = ct*16 + (lane&15).
__global__ void prep_kernel(const float* __restrict__ Wp, const float* __restrict__ Wb,
                            const float* __restrict__ bb, const float* __restrict__ W_ih,
                            const float* __restrict__ W_hh, const float* __restrict__ Wl_ih,
                            const float* __restrict__ Wl_hh, const float* __restrict__ Ws,
                            bf16* __restrict__ Wpf, bf16* __restrict__ W2f,
                            bf16* __restrict__ Wihf, bf16* __restrict__ Whhf,
                            bf16* __restrict__ Wlihf, bf16* __restrict__ Wlhhf,
                            bf16* __restrict__ Wsf) {
  const int total = WPF_N + W2F_N + WIHF_N + WHHF_N + WLIHF_N + WLHHF_N + WSF_N;
  for (int idx = blockIdx.x * blockDim.x + threadIdx.x; idx < total;
       idx += gridDim.x * blockDim.x) {
    int t0 = idx;
    if (t0 < WPF_N) {                         // Wp (64x32): frag = t*2 + ct
      int j = t0 & 15, lane = (t0 >> 4) & 31, f = t0 >> 9;
      int ct = f & 1, t = f >> 1;
      int kb = 16 * (lane >> 4) + j, col = ct * 16 + (lane & 15);
      Wpf[t0] = (bf16)Wp[(32 * t + kb) * DD + col];
    } else if ((t0 -= WPF_N) < W2F_N) {       // W2 = Wb reshaped (512x32) ++ bb (32x32)
      int j = t0 & 15, lane = (t0 >> 4) & 31, f = t0 >> 9;
      int ct = f & 1, t = f >> 1;             // t in [0,16]; t==16 -> bias matrix
      int kb = 16 * (lane >> 4) + j, col = ct * 16 + (lane & 15);
      W2f[t0] = (t < 16) ? (bf16)Wb[(32 * t + kb) * DD + col]
                         : (bf16)bb[kb * DD + col];
    } else if ((t0 -= W2F_N) < WIHF_N) {      // W_ih^T (32x96): frag = ct (6)
      int j = t0 & 15, lane = (t0 >> 4) & 31, ct = t0 >> 9;
      int kb = 16 * (lane >> 4) + j, g = ct * 16 + (lane & 15);
      Wihf[t0] = (bf16)W_ih[g * DD + kb];
    } else if ((t0 -= WIHF_N) < WHHF_N) {     // W_hh^T
      int j = t0 & 15, lane = (t0 >> 4) & 31, ct = t0 >> 9;
      int kb = 16 * (lane >> 4) + j, g = ct * 16 + (lane & 15);
      Whhf[t0] = (bf16)W_hh[g * DD + kb];
    } else if ((t0 -= WHHF_N) < WLIHF_N) {    // Wl_ih^T (128x256): frag = t*16 + ct
      int j = t0 & 15, lane = (t0 >> 4) & 31, f = t0 >> 9;
      int ct = f & 15, t = f >> 4;
      int kb = 16 * (lane >> 4) + j, g = ct * 16 + (lane & 15);
      Wlihf[t0] = (bf16)Wl_ih[g * 128 + 32 * t + kb];
    } else if ((t0 -= WLIHF_N) < WLHHF_N) {   // Wl_hh^T (64x256): frag = t*16 + ct
      int j = t0 & 15, lane = (t0 >> 4) & 31, f = t0 >> 9;
      int ct = f & 15, t = f >> 4;
      int kb = 16 * (lane >> 4) + j, g = ct * 16 + (lane & 15);
      Wlhhf[t0] = (bf16)Wl_hh[g * 64 + 32 * t + kb];
    } else {                                  // Ws (128x512): frag = t*32 + ct
      t0 -= WLHHF_N;
      int j = t0 & 15, lane = (t0 >> 4) & 31, f = t0 >> 9;
      int ct = f & 31, t = f >> 5;
      int kb = 16 * (lane >> 4) + j, col = ct * 16 + (lane & 15);
      Wsf[t0] = (bf16)Ws[(32 * t + kb) * DHH + col];
    }
  }
}

// node2mol is sorted: mol_start[m] = lower_bound(node2mol, m), mol_start[MM] = NN
__global__ void molstart_kernel(const int* __restrict__ node2mol, int* __restrict__ mol_start) {
  int m = blockIdx.x * blockDim.x + threadIdx.x;
  if (m > MM) return;
  int lo = 0, hi = NN;
  while (lo < hi) {
    int mid = (lo + hi) >> 1;
    if (node2mol[mid] < m) lo = mid + 1; else hi = mid;
  }
  mol_start[m] = lo;
}

// ------------------------------------------------- x = relu(node_attr@Wp+bp)
__global__ void proj_kernel(const float* __restrict__ node_attr, const bf16* __restrict__ Wpf,
                            const float* __restrict__ bp, float* __restrict__ h,
                            float* __restrict__ agg0) {
  int wid = (blockIdx.x * blockDim.x + threadIdx.x) >> 5;
  if (wid >= NN / 16) return;
  int lane = threadIdx.x & 31, half = lane >> 4, n = lane & 15;
  int row0 = wid * 16;
  int arow = row0 + n;
  v8f acc[2];
#pragma unroll
  for (int ct = 0; ct < 2; ++ct) {
    float bv = bp[ct * 16 + n];
#pragma unroll
    for (int r = 0; r < 8; ++r) acc[ct][r] = bv;
  }
#pragma unroll
  for (int t = 0; t < 2; ++t) {           // K = 64 -> two 32-chunks
    v16bf a;
#pragma unroll
    for (int j = 0; j < 16; ++j) a[j] = (bf16)node_attr[arow * 64 + 32 * t + kA(j, half)];
#pragma unroll
    for (int ct = 0; ct < 2; ++ct)
      acc[ct] = wmma_bf16(a, load_frag(Wpf, t * 2 + ct, lane), acc[ct]);
  }
#pragma unroll
  for (int ct = 0; ct < 2; ++ct)
#pragma unroll
    for (int r = 0; r < 8; ++r) {
      int row = row0 + r + 8 * half;
      float v = fmaxf(acc[ct][r], 0.0f);
      h[row * DD + ct * 16 + n] = v;
      agg0[row * DD + ct * 16 + n] = v;
    }
}

__global__ void minit_kernel(const float* __restrict__ b_conv, float* __restrict__ mbuf) {
  for (int idx = blockIdx.x * blockDim.x + threadIdx.x; idx < NN * DD;
       idx += gridDim.x * blockDim.x)
    mbuf[idx] = b_conv[idx & 31];
}

// ------------------------- NNConv, fused: msg = outer(edge_attr, x[src]) @ W2
// W2f (bf16, fragment order, bias appended) is DMA'd into LDS with the TDM.
__global__ void nnconv_kernel(const float* __restrict__ edge_attr, const int* __restrict__ edge_src,
                              const int* __restrict__ edge_dst, const bf16* __restrict__ W2f,
                              const float* __restrict__ x, float* __restrict__ mbuf) {
  __shared__ __align__(32) bf16 sW2f[W2F_N];
#if defined(__HIP_DEVICE_COMPILE__) && \
    __has_builtin(__builtin_amdgcn_tensor_load_to_lds) && \
    __has_builtin(__builtin_amdgcn_s_wait_tensorcnt)
  if ((threadIdx.x >> 5) == 0) {          // wave 0 issues one TDM descriptor
    typedef unsigned int v4u __attribute__((ext_vector_type(4)));
    typedef int v8i __attribute__((ext_vector_type(8)));
    typedef int v4i __attribute__((ext_vector_type(4)));
    unsigned long long ga = (unsigned long long)(size_t)W2f;
    unsigned int lds = (unsigned int)(size_t)&sW2f[0];
    const unsigned int n8 = W2F_N * 2u / 8u;   // 4352 8-byte units
    v4u g0;
    g0[0] = 1u;                                 // count=1, load, no gather
    g0[1] = lds;                                // lds_addr
    g0[2] = (unsigned int)(ga & 0xFFFFFFFFu);   // global_addr[31:0]
    g0[3] = (unsigned int)((ga >> 32) & 0x01FFFFFFu) | (2u << 30);  // addr[56:32] | type=2
    v8i g1;
    g1[0] = 3 << 16;                            // data_size = 8B, no multicast
    g1[1] = (int)((n8 & 0xFFFFu) << 16);        // tensor_dim0[15:0]
    g1[2] = (int)((n8 >> 16) | (1u << 16));     // tensor_dim0[31:16] | tensor_dim1[15:0]
    g1[3] = (int)(n8 << 16);                    // tensor_dim1[31:16] | tile_dim0
    g1[4] = 1;                                  // tile_dim1=1, tile_dim2=0
    g1[5] = (int)n8;                            // tensor_dim0_stride[31:0]
    g1[6] = 0;
    g1[7] = 0;
    v4i gz;
    gz[0] = 0; gz[1] = 0; gz[2] = 0; gz[3] = 0;
    v8i gz8;
    gz8[0] = 0; gz8[1] = 0; gz8[2] = 0; gz8[3] = 0;
    gz8[4] = 0; gz8[5] = 0; gz8[6] = 0; gz8[7] = 0;
    // 6-arg toolchain signature: (g0, g1, g2, g3, extra_g, cpol)
    __builtin_amdgcn_tensor_load_to_lds(g0, g1, gz, gz, gz8, 0);
    __builtin_amdgcn_s_wait_tensorcnt(0);
  }
#else
  for (int i = threadIdx.x; i < W2F_N * 2 / 8; i += blockDim.x)
    ((unsigned long long*)sW2f)[i] = ((const unsigned long long*)W2f)[i];
#endif
  __syncthreads();
  int wid = (blockIdx.x * blockDim.x + threadIdx.x) >> 5;
  if (wid >= EE / 16) return;
  int lane = threadIdx.x & 31, half = lane >> 4, n = lane & 15;
  int e = wid * 16 + n;                    // this lane's A-row edge
  int src = edge_src[e];
  float ea[16], xsp[16];
#pragma unroll
  for (int k = 0; k < 16; ++k) ea[k] = edge_attr[e * 16 + k];
#pragma unroll
  for (int j = 0; j < 16; ++j) xsp[j] = x[src * DD + kA(j, half)];  // pre-permuted x[src]
  v8f acc[2];
#pragma unroll
  for (int ct = 0; ct < 2; ++ct)
#pragma unroll
    for (int r = 0; r < 8; ++r) acc[ct][r] = 0.0f;
#pragma unroll
  for (int t = 0; t < 16; ++t) {          // K = 512 = 16 chunks; chunk t <-> k=t
    v16bf a;
#pragma unroll
    for (int j = 0; j < 16; ++j) a[j] = (bf16)(ea[t] * xsp[j]);
#pragma unroll
    for (int ct = 0; ct < 2; ++ct) {
      v16bf b = *(const v16bf*)(sW2f + ((size_t)(t * 2 + ct) * 32 + lane) * 16);
      acc[ct] = wmma_bf16(a, b, acc[ct]);
    }
  }
  {                                        // + x[src] @ reshape(bb, 32,32)
    v16bf a;
#pragma unroll
    for (int j = 0; j < 16; ++j) a[j] = (bf16)xsp[j];
#pragma unroll
    for (int ct = 0; ct < 2; ++ct) {
      v16bf b = *(const v16bf*)(sW2f + ((size_t)(32 + ct) * 32 + lane) * 16);
      acc[ct] = wmma_bf16(a, b, acc[ct]);
    }
  }
#pragma unroll
  for (int r = 0; r < 8; ++r) {            // scatter-sum to dst nodes
    int er = wid * 16 + r + 8 * half;
    int d = edge_dst[er];
#pragma unroll
    for (int ct = 0; ct < 2; ++ct)
      atomicAdd(&mbuf[d * DD + ct * 16 + n], acc[ct][r]);
  }
}

// --------------------------------------------------------------- GRU cell
__global__ void gru_kernel(const float* __restrict__ mbuf, const bf16* __restrict__ Wihf,
                           const bf16* __restrict__ Whhf, const float* __restrict__ b_ih,
                           const float* __restrict__ b_hh, float* __restrict__ h) {
  int wid = (blockIdx.x * blockDim.x + threadIdx.x) >> 5;
  if (wid >= NN / 16) return;
  int lane = threadIdx.x & 31, half = lane >> 4, n = lane & 15;
  int row0 = wid * 16;
  int arow = row0 + n;
  v16bf ax, ah;
#pragma unroll
  for (int j = 0; j < 16; ++j) {
    ax[j] = (bf16)fmaxf(mbuf[arow * DD + kA(j, half)], 0.0f);  // xin = relu(m)
    ah[j] = (bf16)h[arow * DD + kA(j, half)];
  }
  v8f gi[6], gh[6];
#pragma unroll
  for (int ct = 0; ct < 6; ++ct) {
    float bi = b_ih[ct * 16 + n], bh = b_hh[ct * 16 + n];
#pragma unroll
    for (int r = 0; r < 8; ++r) { gi[ct][r] = bi; gh[ct][r] = bh; }
  }
#pragma unroll
  for (int ct = 0; ct < 6; ++ct) {        // K = 32, one WMMA each
    gi[ct] = wmma_bf16(ax, load_frag(Wihf, ct, lane), gi[ct]);
    gh[ct] = wmma_bf16(ah, load_frag(Whhf, ct, lane), gh[ct]);
  }
#pragma unroll
  for (int ct = 0; ct < 2; ++ct)          // hidden col tiles
#pragma unroll
    for (int r = 0; r < 8; ++r) {
      int row = row0 + r + 8 * half;
      int col = ct * 16 + n;
      float hold = h[row * DD + col];
      float rg = sigm(gi[ct][r] + gh[ct][r]);
      float zg = sigm(gi[2 + ct][r] + gh[2 + ct][r]);
      float ng = tanhf(gi[4 + ct][r] + rg * gh[4 + ct][r]);
      h[row * DD + col] = (1.0f - zg) * ng + zg * hold;
    }
}

__global__ void feat_kernel(const float* __restrict__ h, const float* __restrict__ agg0,
                            float* __restrict__ feat) {
  for (int idx = blockIdx.x * blockDim.x + threadIdx.x; idx < NN * 64;
       idx += gridDim.x * blockDim.x) {
    int row = idx >> 6, c = idx & 63;
    feat[idx] = (c < DD) ? h[row * DD + c] : agg0[row * DD + (c - DD)];
  }
}

__global__ void zero_kernel(float* __restrict__ p, int count) {
  for (int idx = blockIdx.x * blockDim.x + threadIdx.x; idx < count;
       idx += gridDim.x * blockDim.x)
    p[idx] = 0.0f;
}

// ----------------------------------------------------------- Set2Set LSTM
__global__ void lstm_kernel(const float* __restrict__ q_star, const bf16* __restrict__ Wlihf,
                            const bf16* __restrict__ Wlhhf, const float* __restrict__ bl_ih,
                            const float* __restrict__ bl_hh, float* __restrict__ hL,
                            float* __restrict__ cL) {
  int wid = (blockIdx.x * blockDim.x + threadIdx.x) >> 5;
  if (wid >= (MM + 15) / 16) return;
  int lane = threadIdx.x & 31, half = lane >> 4, n = lane & 15;
  int row0 = wid * 16;
  int arow = row0 + n;
  bool aok = arow < MM;
  v8f acc[16];
#pragma unroll
  for (int ct = 0; ct < 16; ++ct) {
    float bv = bl_ih[ct * 16 + n] + bl_hh[ct * 16 + n];
#pragma unroll
    for (int r = 0; r < 8; ++r) acc[ct][r] = bv;
  }
#pragma unroll
  for (int t = 0; t < 4; ++t) {           // q_star @ Wl_ih^T, K=128
    v16bf a;
#pragma unroll
    for (int j = 0; j < 16; ++j)
      a[j] = aok ? (bf16)q_star[arow * 128 + 32 * t + kA(j, half)] : (bf16)0.0f;
#pragma unroll
    for (int ct = 0; ct < 16; ++ct)
      acc[ct] = wmma_bf16(a, load_frag(Wlihf, t * 16 + ct, lane), acc[ct]);
  }
#pragma unroll
  for (int t = 0; t < 2; ++t) {           // hL @ Wl_hh^T, K=64
    v16bf a;
#pragma unroll
    for (int j = 0; j < 16; ++j)
      a[j] = aok ? (bf16)hL[arow * 64 + 32 * t + kA(j, half)] : (bf16)0.0f;
#pragma unroll
    for (int ct = 0; ct < 16; ++ct)
      acc[ct] = wmma_bf16(a, load_frag(Wlhhf, t * 16 + ct, lane), acc[ct]);
  }
#pragma unroll
  for (int ct = 0; ct < 4; ++ct)          // gates i,f,g,o = tiles ct, 4+ct, 8+ct, 12+ct
#pragma unroll
    for (int r = 0; r < 8; ++r) {
      int row = row0 + r + 8 * half;
      if (row < MM) {
        int col = ct * 16 + n;
        float cold = cL[row * 64 + col];
        float iv = sigm(acc[ct][r]);
        float fv = sigm(acc[4 + ct][r]);
        float gv = tanhf(acc[8 + ct][r]);
        float ov = sigm(acc[12 + ct][r]);
        float cn = fv * cold + iv * gv;
        cL[row * 64 + col] = cn;
        hL[row * 64 + col] = ov * tanhf(cn);
      }
    }
}

// ----------------------------- segment softmax attention + readout (1 wave/mol)
__global__ void attn_kernel(const float* __restrict__ feat, const int* __restrict__ mol_start,
                            const float* __restrict__ hL, float* __restrict__ q_star,
                            float* __restrict__ ebuf) {
  int m = blockIdx.x;
  if (m >= MM) return;
  int lane = threadIdx.x;
  int s0 = mol_start[m], s1 = mol_start[m + 1];
  float q0 = hL[m * 64 + lane];
  float q1 = hL[m * 64 + 32 + lane];
  float mx = -3.4e38f;
  for (int i = s0; i < s1; ++i) {
    float p = feat[i * 64 + lane] * q0 + feat[i * 64 + 32 + lane] * q1;
#pragma unroll
    for (int off = 16; off > 0; off >>= 1) p += __shfl_xor(p, off, 32);
    ebuf[i] = p;
    mx = fmaxf(mx, p);
  }
  float s = 0.0f, r0 = 0.0f, r1 = 0.0f;
  for (int i = s0; i < s1; ++i) {
    float ex = __expf(ebuf[i] - mx);
    s += ex;
    r0 += ex * feat[i * 64 + lane];
    r1 += ex * feat[i * 64 + 32 + lane];
  }
  float inv = (s > 0.0f) ? 1.0f / s : 0.0f;
  q_star[m * 128 + lane] = q0;
  q_star[m * 128 + 32 + lane] = q1;
  q_star[m * 128 + 64 + lane] = r0 * inv;
  q_star[m * 128 + 96 + lane] = r1 * inv;
}

// -------------------------- z2 = prelu(q_star @ Ws + bs); segsum over mol2rxn
__global__ void sparsify_kernel(const float* __restrict__ q_star, const bf16* __restrict__ Wsf,
                                const float* __restrict__ bs, const float* __restrict__ prelu_a,
                                const int* __restrict__ mol2rxn, float* __restrict__ out) {
  int wid = (blockIdx.x * blockDim.x + threadIdx.x) >> 5;
  const int nrt = (MM + 15) / 16;          // 63 row tiles, 32 col tiles
  if (wid >= nrt * 32) return;
  int rt = wid >> 5, ct = wid & 31;
  int lane = threadIdx.x & 31, half = lane >> 4, n = lane & 15;
  int row0 = rt * 16;
  int arow = row0 + n;
  bool aok = arow < MM;
  v8f acc;
  float bv = bs[ct * 16 + n];
#pragma unroll
  for (int r = 0; r < 8; ++r) acc[r] = bv;
#pragma unroll
  for (int t = 0; t < 4; ++t) {            // K = 128
    v16bf a;
#pragma unroll
    for (int j = 0; j < 16; ++j)
      a[j] = aok ? (bf16)q_star[arow * 128 + 32 * t + kA(j, half)] : (bf16)0.0f;
    acc = wmma_bf16(a, load_frag(Wsf, t * 32 + ct, lane), acc);
  }
  float pa = prelu_a[0];
#pragma unroll
  for (int r = 0; r < 8; ++r) {
    int row = row0 + r + 8 * half;
    if (row < MM) {
      float v = acc[r];
      v = (v >= 0.0f) ? v : pa * v;
      atomicAdd(&out[mol2rxn[row] * DHH + ct * 16 + n], v);
    }
  }
}

// ============================================================== launcher
extern "C" void kernel_launch(void* const* d_in, const int* in_sizes, int n_in,
                              void* d_out, int out_size, void* d_ws, size_t ws_size,
                              hipStream_t stream) {
  const float* node_attr = (const float*)d_in[0];
  const float* edge_attr = (const float*)d_in[1];
  const int*   edge_src  = (const int*)d_in[2];
  const int*   edge_dst  = (const int*)d_in[3];
  const int*   node2mol  = (const int*)d_in[4];
  const int*   mol2rxn   = (const int*)d_in[5];
  const float* Wp   = (const float*)d_in[6];
  const float* bp   = (const float*)d_in[7];
  const float* Wb   = (const float*)d_in[8];
  const float* bb   = (const float*)d_in[9];
  const float* b_conv = (const float*)d_in[10];
  const float* W_ih = (const float*)d_in[11];
  const float* W_hh = (const float*)d_in[12];
  const float* b_ih = (const float*)d_in[13];
  const float* b_hh = (const float*)d_in[14];
  const float* Wl_ih = (const float*)d_in[15];
  const float* Wl_hh = (const float*)d_in[16];
  const float* bl_ih = (const float*)d_in[17];
  const float* bl_hh = (const float*)d_in[18];
  const float* Ws   = (const float*)d_in[19];
  const float* bs   = (const float*)d_in[20];
  const float* prelu_a = (const float*)d_in[21];
  float* out = (float*)d_out;

  char* ws = (char*)d_ws;
  size_t off = 0;
  auto alloc = [&](size_t bytes) -> void* {
    void* p = ws + off;
    off = (off + bytes + 255) & ~(size_t)255;
    return p;
  };
  float* h      = (float*)alloc((size_t)NN * DD * 4);
  float* agg0   = (float*)alloc((size_t)NN * DD * 4);
  float* mbuf   = (float*)alloc((size_t)NN * DD * 4);
  float* feat   = (float*)alloc((size_t)NN * 64 * 4);
  float* ebuf   = (float*)alloc((size_t)NN * 4);
  float* hL     = (float*)alloc((size_t)MM * 64 * 4);
  float* cL     = (float*)alloc((size_t)MM * 64 * 4);
  float* q_star = (float*)alloc((size_t)MM * 128 * 4);
  int*   mol_start = (int*)alloc((size_t)(MM + 1) * 4);
  bf16* Wpf   = (bf16*)alloc(WPF_N * 2);
  bf16* W2f   = (bf16*)alloc(W2F_N * 2);
  bf16* Wihf  = (bf16*)alloc(WIHF_N * 2);
  bf16* Whhf  = (bf16*)alloc(WHHF_N * 2);
  bf16* Wlihf = (bf16*)alloc(WLIHF_N * 2);
  bf16* Wlhhf = (bf16*)alloc(WLHHF_N * 2);
  bf16* Wsf   = (bf16*)alloc(WSF_N * 2);

  prep_kernel<<<548, 256, 0, stream>>>(Wp, Wb, bb, W_ih, W_hh, Wl_ih, Wl_hh, Ws,
                                        Wpf, W2f, Wihf, Whhf, Wlihf, Wlhhf, Wsf);
  molstart_kernel<<<(MM + 1 + 255) / 256, 256, 0, stream>>>(node2mol, mol_start);

  // x = relu(node_attr @ Wp + bp); 1250 row tiles, 8 waves/block
  proj_kernel<<<(NN / 16 + 7) / 8, 256, 0, stream>>>(node_attr, Wpf, bp, h, agg0);

  for (int step = 0; step < 3; ++step) {
    minit_kernel<<<1250, 256, 0, stream>>>(b_conv, mbuf);
    nnconv_kernel<<<(EE / 16 + 7) / 8, 256, 0, stream>>>(edge_attr, edge_src, edge_dst,
                                                          W2f, h, mbuf);
    gru_kernel<<<(NN / 16 + 7) / 8, 256, 0, stream>>>(mbuf, Wihf, Whhf, b_ih, b_hh, h);
  }

  feat_kernel<<<2500, 256, 0, stream>>>(h, agg0, feat);
  zero_kernel<<<256, 256, 0, stream>>>(hL, MM * 64);
  zero_kernel<<<256, 256, 0, stream>>>(cL, MM * 64);
  zero_kernel<<<512, 256, 0, stream>>>(q_star, MM * 128);

  for (int it = 0; it < 3; ++it) {
    lstm_kernel<<<8, 256, 0, stream>>>(q_star, Wlihf, Wlhhf, bl_ih, bl_hh, hL, cL);
    attn_kernel<<<MM, 32, 0, stream>>>(feat, mol_start, hL, q_star, ebuf);
  }

  zero_kernel<<<500, 256, 0, stream>>>(out, RR * DHH);
  sparsify_kernel<<<(((MM + 15) / 16) * 32 + 7) / 8, 256, 0, stream>>>(q_star, Wsf, bs,
                                                                        prelu_a, mol2rxn, out);
}
